// TransformerBlock_80685255623338
// MI455X (gfx1250) — matrix operand order repaired and checked by reading, not compiled, not run on hardware
//
#include <hip/hip_runtime.h>
#include <hip/hip_bf16.h>

// ---------------------------------------------------------------------------
// MI455X (gfx1250) transformer block, bf16 WMMA everywhere.
// B=4, T=1024, DIM=2048, H=16, HKV=4, DH=128, GROUP=4, HIDDEN=8192, M=4096.
// GEMMs: LDS double-buffered; A tile via GLOBAL_LOAD_ASYNC_TO_LDS_B128
// (ASYNCcnt), B tile via the Tensor Data Mover (TENSORcnt).
// ---------------------------------------------------------------------------

typedef __attribute__((ext_vector_type(16))) __bf16 v16bf;
typedef __attribute__((ext_vector_type(8)))  __bf16 v8bf;
typedef __attribute__((ext_vector_type(8)))  float  v8f;
typedef __attribute__((ext_vector_type(4)))  unsigned int u32x4;
typedef __attribute__((ext_vector_type(8)))  int          i32x8;
typedef __attribute__((ext_vector_type(4)))  int          i32x4;

#define DIMC    2048
#define T_SEQ   1024
#define B_SZ    4
#define H_Q     16
#define H_KV    4
#define DHEAD   128
#define GRP     4
#define HID     8192
#define M_ROWS  (B_SZ * T_SEQ)   // 4096

__device__ __forceinline__ v16bf cat8(v8bf lo, v8bf hi) {
  return __builtin_shufflevector(lo, hi, 0,1,2,3,4,5,6,7,8,9,10,11,12,13,14,15);
}

// Low 32 bits of a generic pointer to __shared__ are the LDS byte address
// (ISA 10.2, LDS aperture).
__device__ __forceinline__ unsigned lds_addr32(const void* p) {
  return (unsigned)(unsigned long long)p;
}

// GLOBAL_LOAD_ASYNC_TO_LDS_B128: per-lane 16B global->LDS, tracked by ASYNCcnt.
__device__ __forceinline__ void async_copy_b128(const void* gptr, void* lptr) {
  unsigned loff = lds_addr32(lptr);
  asm volatile("global_load_async_to_lds_b128 %0, %1, off"
               :: "v"(loff), "v"((unsigned long long)gptr)
               : "memory");
}
__device__ __forceinline__ void wait_async_le1() {
  asm volatile("s_wait_asynccnt 0x1" ::: "memory");
}
__device__ __forceinline__ void wait_async_le0() {
  asm volatile("s_wait_asynccnt 0x0" ::: "memory");
}

// Tensor Data Mover: DMA a [rows=128][k=32] bf16 tile of Bt (row pitch K)
// into contiguous LDS.  D# packed per CDNA5 ISA 8.3/8.4 (count=1, type=2,
// data_size=2B, tensor_dim0=K, tensor_dim0_stride=K, tile 32x128).
__device__ __forceinline__ void tdm_load_Btile(const __bf16* gsrc, void* ldst,
                                               unsigned K, unsigned N) {
  unsigned long long ga = (unsigned long long)gsrc;
  u32x4 g0;
  g0[0] = 1u;                                           // count=1, user mode
  g0[1] = lds_addr32(ldst);                             // lds_addr
  g0[2] = (unsigned)ga;                                 // global_addr[31:0]
  g0[3] = (unsigned)((ga >> 32) & 0x01FFFFFFu) | (2u << 30);  // [56:32] | type=2
  i32x8 g1;
  g1[0] = (int)(1u << 16);                              // data_size=1 (2 bytes)
  g1[1] = (int)((K & 0xFFFFu) << 16);                   // tensor_dim0[15:0]
  g1[2] = (int)(((K >> 16) & 0xFFFFu) | ((N & 0xFFFFu) << 16));   // dim0 hi | dim1 lo
  g1[3] = (int)(((N >> 16) & 0xFFFFu) | (32u << 16));   // dim1 hi | tile_dim0=32
  g1[4] = 128;                                          // tile_dim1=128
  g1[5] = (int)K;                                       // tensor_dim0_stride[31:0]
  g1[6] = 0;                                            // stride hi | dim1_stride lo
  g1[7] = 0;
  i32x4 z4 = {};
  i32x8 z8 = {};
  // 6-arg toolchain form: (g0, g1, g2, g3, extra, cpol)
  __builtin_amdgcn_tensor_load_to_lds(g0, g1, z4, z4, z8, 0);
}

// -------------------------------------------------------------------------
// Weight prep: f32 [K,N] -> bf16 [N,K] (transposed -> K-contiguous B reads).
// -------------------------------------------------------------------------
__global__ __launch_bounds__(256) void cvt_transpose_f32_bf16(
    const float* __restrict__ W, __bf16* __restrict__ Wt, int K, int N) {
  __shared__ __bf16 tile[32][33];
  int tx = threadIdx.x, ty = threadIdx.y;            // blockDim (32, 8)
  int n0 = blockIdx.x * 32, k0 = blockIdx.y * 32;
  #pragma unroll
  for (int r = 0; r < 32; r += 8)
    tile[ty + r][tx] = (__bf16)W[(long)(k0 + ty + r) * N + (n0 + tx)];
  __syncthreads();
  #pragma unroll
  for (int r = 0; r < 32; r += 8)
    Wt[(long)(n0 + ty + r) * K + (k0 + tx)] = tile[tx][ty + r];
}

// -------------------------------------------------------------------------
// RMSNorm row kernel (one 256-thread block per token row), bf16 output.
// -------------------------------------------------------------------------
__global__ __launch_bounds__(256) void rmsnorm_bf16(
    const float* __restrict__ x, const float* __restrict__ w,
    __bf16* __restrict__ out) {
  long row = blockIdx.x;
  const float* xr = x + row * DIMC;
  float ss = 0.f;
  for (int i = threadIdx.x; i < DIMC; i += 256) { float v = xr[i]; ss += v * v; }
  #pragma unroll
  for (int off = 16; off; off >>= 1) ss += __shfl_xor(ss, off, 32);
  __shared__ float red[8];
  int wave = threadIdx.x >> 5, lane = threadIdx.x & 31;
  if (lane == 0) red[wave] = ss;
  __syncthreads();
  if (wave == 0) {
    float v = (lane < 8) ? red[lane] : 0.f;
    #pragma unroll
    for (int off = 4; off; off >>= 1) v += __shfl_xor(v, off, 32);
    if (lane == 0) red[0] = v;
  }
  __syncthreads();
  float scale = rsqrtf(red[0] / (float)DIMC + 1e-5f);
  for (int i = threadIdx.x; i < DIMC; i += 256)
    out[row * DIMC + i] = (__bf16)(xr[i] * scale * w[i]);
}

// -------------------------------------------------------------------------
// GEMM: C[M,N](f32) (+)= A[M,K](bf16) * Bt[N,K](bf16).  Block tile 64x128,
// 8 waves = 4(M)x2(N), each wave 16x64 (4 accumulators).  LDS double buffer:
// A tile (64x32, 4KB) via per-thread async b128 copies; B tile (128x32, 8KB)
// via one TDM descriptor issued by wave 0.  copy(i+1) overlaps WMMA(i).
// -------------------------------------------------------------------------
template <int ACC>
__global__ __launch_bounds__(256) void gemm_bf16(
    const __bf16* __restrict__ A, const __bf16* __restrict__ Bt,
    float* __restrict__ C, int M, int N, int K) {
  __shared__ __attribute__((aligned(32))) __bf16 Abuf[2][64][32];    // 2x4KB
  __shared__ __attribute__((aligned(32))) __bf16 Bbuf[2][128][32];   // 2x8KB
  int tid = threadIdx.x;
  int lane = tid & 31, wave = tid >> 5;
  int wm = wave >> 1, wn = wave & 1;
  long mblk = (long)blockIdx.x * 64;
  long nblk = (long)blockIdx.y * 128;
  int lr    = lane & 15;
  int koffA = (lane >> 4) * 16;   // A-operand K chunk base
  int koffB = (lane >> 4) * 16;   // B-operand K chunk base
  int rbase = (lane >> 4) * 8;

  // A-tile cooperative copy mapping: thread -> (row, 8-elem chunk)
  int car = tid >> 2;            // 0..63
  int cac = (tid & 3) * 8;       // 0,8,16,24
  const __bf16* agsrc = A + (mblk + car) * (long)K + cac;
  const __bf16* bgsrc = Bt + nblk * (long)K;

  int nsteps = K >> 5;
  // prologue: stage k-step 0 into buffer 0
  async_copy_b128(agsrc, &Abuf[0][car][cac]);
  if (wave == 0) tdm_load_Btile(bgsrc, &Bbuf[0][0][0], (unsigned)K, (unsigned)N);

  v8f acc[4] = {};
  for (int i = 0; i < nsteps; ++i) {
    int cur = i & 1;
    if (i + 1 < nsteps) {
      int k2 = (i + 1) << 5;
      async_copy_b128(agsrc + k2, &Abuf[1 - cur][car][cac]);
      if (wave == 0)
        tdm_load_Btile(bgsrc + k2, &Bbuf[1 - cur][0][0], (unsigned)K, (unsigned)N);
      wait_async_le1();                                   // my step-i A copy done
      if (wave == 0) __builtin_amdgcn_s_wait_tensorcnt(1); // step-i B DMA done
    } else {
      wait_async_le0();
      if (wave == 0) __builtin_amdgcn_s_wait_tensorcnt(0);
    }
    __syncthreads();                                       // tile visible to all

    v8bf alo = *(const v8bf*)(&Abuf[cur][wm * 16 + lr][koffA]);
    v8bf ahi = *(const v8bf*)(&Abuf[cur][wm * 16 + lr][8 + koffA]);
    v16bf av = cat8(alo, ahi);
    #pragma unroll
    for (int j = 0; j < 4; ++j) {
      v16bf bv = *(const v16bf*)(&Bbuf[cur][wn * 64 + j * 16 + lr][koffB]);
      acc[j] = __builtin_amdgcn_wmma_f32_16x16x32_bf16(
          false, av, false, bv, (short)0, acc[j], false, false);
    }
    __syncthreads();                  // all reads done before buf is re-filled
  }

  long m0 = mblk + wm * 16, n0 = nblk + wn * 64;
  #pragma unroll
  for (int j = 0; j < 4; ++j) {
    long col = n0 + j * 16 + lr;
    #pragma unroll
    for (int r = 0; r < 8; ++r) {
      long idx = (m0 + rbase + r) * (long)N + col;
      if (ACC) C[idx] += acc[j][r]; else C[idx] = acc[j][r];
    }
  }
}

// -------------------------------------------------------------------------
// RoPE + f32->bf16.  src/dst: [M, NH*DH]; one thread per (row, head, freq).
// -------------------------------------------------------------------------
__global__ void rope_cvt(const float* __restrict__ src, __bf16* __restrict__ dst,
                         int NH, long total) {
  long idx = (long)blockIdx.x * blockDim.x + threadIdx.x;
  if (idx >= total) return;
  int  i   = (int)(idx % 64);
  long tmp = idx / 64;
  int  h   = (int)(tmp % NH);
  long row = tmp / NH;
  int  t   = (int)(row % T_SEQ);
  float inv = __powf(10000.f, -(float)i / 64.f);
  float s, c;
  __sincosf((float)t * inv, &s, &c);
  long base = row * (long)(NH * DHEAD) + (long)h * DHEAD;
  float x1 = src[base + i], x2 = src[base + 64 + i];
  dst[base + i]      = (__bf16)(x1 * c - x2 * s);
  dst[base + 64 + i] = (__bf16)(x1 * s + x2 * c);
}

// V: f32 [B*T, HKV*DH] -> bf16 transposed [B, HKV, DH, T]  (P*V B-operand
// becomes key-contiguous).
__global__ void v_cvt_transpose(const float* __restrict__ v, __bf16* __restrict__ vt) {
  long idx = (long)blockIdx.x * blockDim.x + threadIdx.x;  // M_ROWS*H_KV*DHEAD
  int  d   = (int)(idx % DHEAD);
  long tmp = idx / DHEAD;
  int  kh  = (int)(tmp % H_KV);
  long row = tmp / H_KV;
  int  t   = (int)(row % T_SEQ);
  int  b   = (int)(row / T_SEQ);
  vt[(((long)b * H_KV + kh) * DHEAD + d) * T_SEQ + t] =
      (__bf16)v[row * (H_KV * DHEAD) + kh * DHEAD + d];
}

// -------------------------------------------------------------------------
// Flash-style causal GQA attention.  One wave = 16 query rows of one head;
// 32 keys per iteration so P (16x32) is exactly one WMMA A-operand.
// grid = (T/128, H, B), block = 256 (8 waves).
// -------------------------------------------------------------------------
__global__ __launch_bounds__(256) void attn_kernel(
    const __bf16* __restrict__ q, const __bf16* __restrict__ k,
    const __bf16* __restrict__ vt, float* __restrict__ y) {
  __shared__ __attribute__((aligned(32))) __bf16 plds[8][16][32];
  int lane = threadIdx.x & 31, wave = threadIdx.x >> 5;
  int b = blockIdx.z, h = blockIdx.y;
  int kh = h / GRP;
  int t0 = (blockIdx.x * 8 + wave) * 16;
  int lr = lane & 15;
  int koffA = (lane >> 4) * 16;
  int koffB = (lane >> 4) * 16;
  int rbase = (lane >> 4) * 8;

  const __bf16* qrow  = q  + ((long)b * T_SEQ + t0 + lr) * (H_Q * DHEAD) + h * DHEAD;
  const __bf16* kbase = k  + (long)b * T_SEQ * (H_KV * DHEAD) + kh * DHEAD;
  const __bf16* vbase = vt + ((long)b * H_KV + kh) * DHEAD * (long)T_SEQ;

  v16bf qa[4];
  #pragma unroll
  for (int kk = 0; kk < 4; ++kk) {
    v8bf lo = *(const v8bf*)(qrow + kk * 32 + koffA);
    v8bf hi = *(const v8bf*)(qrow + kk * 32 + 8 + koffA);
    qa[kk] = cat8(lo, hi);
  }

  v8f O[8] = {};
  float mi[8], li[8];
  #pragma unroll
  for (int r = 0; r < 8; ++r) { mi[r] = -1e30f; li[r] = 0.f; }
  const float scale = 0.08838834764831845f;   // 1/sqrt(128)

  for (int s0 = 0; s0 < t0 + 16; s0 += 32) {
    int si0 = s0 + lr;       si0 = si0 < T_SEQ ? si0 : T_SEQ - 1;
    int si1 = s0 + 16 + lr;  si1 = si1 < T_SEQ ? si1 : T_SEQ - 1;
    const __bf16* kr0 = kbase + (long)si0 * (H_KV * DHEAD);
    const __bf16* kr1 = kbase + (long)si1 * (H_KV * DHEAD);
    v8f S0 = {}, S1 = {};
    #pragma unroll
    for (int kk = 0; kk < 4; ++kk) {
      v16bf b0 = *(const v16bf*)(kr0 + kk * 32 + koffB);
      v16bf b1 = *(const v16bf*)(kr1 + kk * 32 + koffB);
      S0 = __builtin_amdgcn_wmma_f32_16x16x32_bf16(false, qa[kk], false, b0,
                                                   (short)0, S0, false, false);
      S1 = __builtin_amdgcn_wmma_f32_16x16x32_bf16(false, qa[kk], false, b1,
                                                   (short)0, S1, false, false);
    }
    float corr[8];
    #pragma unroll
    for (int r = 0; r < 8; ++r) {
      int tq = t0 + rbase + r;
      float s0v = ((s0 + lr)      <= tq) ? S0[r] * scale : -1e30f;
      float s1v = ((s0 + 16 + lr) <= tq) ? S1[r] * scale : -1e30f;
      float mx = fmaxf(s0v, s1v);
      #pragma unroll
      for (int off = 8; off; off >>= 1) mx = fmaxf(mx, __shfl_xor(mx, off, 32));
      float mnew = fmaxf(mi[r], mx);
      float e0 = __expf(s0v - mnew);
      float e1 = __expf(s1v - mnew);
      corr[r] = __expf(mi[r] - mnew);
      float ps = e0 + e1;
      #pragma unroll
      for (int off = 8; off; off >>= 1) ps += __shfl_xor(ps, off, 32);
      li[r] = li[r] * corr[r] + ps;
      mi[r] = mnew;
      plds[wave][rbase + r][lr]      = (__bf16)e0;   // P -> LDS (C->A relayout)
      plds[wave][rbase + r][16 + lr] = (__bf16)e1;
    }
    #pragma unroll
    for (int j = 0; j < 8; ++j)
      #pragma unroll
      for (int r = 0; r < 8; ++r) O[j][r] *= corr[r];

    v8bf plo = *(const v8bf*)(&plds[wave][lr][koffA]);
    v8bf phi = *(const v8bf*)(&plds[wave][lr][8 + koffA]);
    v16bf pv = cat8(plo, phi);
    #pragma unroll
    for (int j = 0; j < 8; ++j) {
      const __bf16* vr = vbase + (long)(j * 16 + lr) * T_SEQ + s0 + koffB;
      v16bf bv = *(const v16bf*)vr;
      O[j] = __builtin_amdgcn_wmma_f32_16x16x32_bf16(false, pv, false, bv,
                                                     (short)0, O[j], false, false);
    }
  }

  #pragma unroll
  for (int j = 0; j < 8; ++j) {
    long col = (long)h * DHEAD + j * 16 + lr;
    #pragma unroll
    for (int r = 0; r < 8; ++r) {
      long row = (long)b * T_SEQ + t0 + rbase + r;
      y[row * (H_Q * DHEAD) + col] = O[j][r] / li[r];
    }
  }
}

// -------------------------------------------------------------------------
// Elementwise helpers.
// -------------------------------------------------------------------------
__global__ void cvt_bf16(const float* __restrict__ src, __bf16* __restrict__ dst, long n) {
  long i = (long)blockIdx.x * blockDim.x + threadIdx.x;
  if (i < n) dst[i] = (__bf16)src[i];
}

__global__ void silu_mul_bf16(const float* __restrict__ h1, const float* __restrict__ h2,
                              __bf16* __restrict__ out, long n) {
  long i = (long)blockIdx.x * blockDim.x + threadIdx.x;
  if (i < n) {
    float a = h1[i], g = h2[i];
    out[i] = (__bf16)((a / (1.f + __expf(-a))) * g);
  }
}

// -------------------------------------------------------------------------
// Host-side orchestration.
// -------------------------------------------------------------------------
extern "C" void kernel_launch(void* const* d_in, const int* in_sizes, int n_in,
                              void* d_out, int out_size, void* d_ws, size_t ws_size,
                              hipStream_t stream) {
  (void)in_sizes; (void)n_in; (void)out_size; (void)ws_size;
  const float* x   = (const float*)d_in[0];
  const float* anw = (const float*)d_in[1];
  const float* fnw = (const float*)d_in[2];
  const float* wq  = (const float*)d_in[3];
  const float* wk  = (const float*)d_in[4];
  const float* wv  = (const float*)d_in[5];
  const float* wo  = (const float*)d_in[6];
  const float* w1  = (const float*)d_in[7];
  const float* w2  = (const float*)d_in[8];
  const float* w3  = (const float*)d_in[9];
  float* out = (float*)d_out;
  char*  ws  = (char*)d_ws;
  const size_t MB = 1ull << 20;

  __bf16* wq_t = (__bf16*)(ws + 0 * MB);     //  8 MB  [2048][2048]
  __bf16* wk_t = (__bf16*)(ws + 8 * MB);     //  2 MB  [512][2048]
  __bf16* wv_t = (__bf16*)(ws + 10 * MB);    //  2 MB  [512][2048]
  __bf16* wo_t = (__bf16*)(ws + 12 * MB);    //  8 MB  [2048][2048]
  __bf16* w1_t = (__bf16*)(ws + 20 * MB);    // 32 MB  [8192][2048]
  __bf16* w2_t = (__bf16*)(ws + 52 * MB);    // 32 MB  [8192][2048]
  __bf16* w3_t = (__bf16*)(ws + 84 * MB);    // 32 MB  [2048][8192]
  __bf16* a_bf = (__bf16*)(ws + 116 * MB);   // 16 MB  (reused as FFN b_bf)
  float*  q_f  = (float*)(ws + 132 * MB);    // 32 MB
  float*  k_f  = (float*)(ws + 164 * MB);    //  8 MB
  float*  v_f  = (float*)(ws + 172 * MB);    //  8 MB
  __bf16* q_bf = (__bf16*)(ws + 180 * MB);   // 16 MB
  __bf16* k_bf = (__bf16*)(ws + 196 * MB);   //  4 MB
  __bf16* vt_bf= (__bf16*)(ws + 200 * MB);   //  4 MB
  float*  y_f  = (float*)(ws + 204 * MB);    // 32 MB
  __bf16* y_bf = (__bf16*)(ws + 236 * MB);   // 16 MB
  float*  h1   = (float*)(ws + 252 * MB);    // 128 MB
  float*  h2   = (float*)(ws + 380 * MB);    // 128 MB
  __bf16* hh   = (__bf16*)(ws + 508 * MB);   // 64 MB

  dim3 tb(32, 8);
  cvt_transpose_f32_bf16<<<dim3(64, 64),  tb, 0, stream>>>(wq, wq_t, 2048, 2048);
  cvt_transpose_f32_bf16<<<dim3(16, 64),  tb, 0, stream>>>(wk, wk_t, 2048, 512);
  cvt_transpose_f32_bf16<<<dim3(16, 64),  tb, 0, stream>>>(wv, wv_t, 2048, 512);
  cvt_transpose_f32_bf16<<<dim3(64, 64),  tb, 0, stream>>>(wo, wo_t, 2048, 2048);
  cvt_transpose_f32_bf16<<<dim3(256, 64), tb, 0, stream>>>(w1, w1_t, 2048, 8192);
  cvt_transpose_f32_bf16<<<dim3(256, 64), tb, 0, stream>>>(w2, w2_t, 2048, 8192);
  cvt_transpose_f32_bf16<<<dim3(64, 256), tb, 0, stream>>>(w3, w3_t, 8192, 2048);

  // attention half
  rmsnorm_bf16<<<M_ROWS, 256, 0, stream>>>(x, anw, a_bf);
  gemm_bf16<0><<<dim3(64, 16), 256, 0, stream>>>(a_bf, wq_t, q_f, M_ROWS, 2048, 2048);
  gemm_bf16<0><<<dim3(64, 4),  256, 0, stream>>>(a_bf, wk_t, k_f, M_ROWS, 512,  2048);
  gemm_bf16<0><<<dim3(64, 4),  256, 0, stream>>>(a_bf, wv_t, v_f, M_ROWS, 512,  2048);
  rope_cvt<<<16384, 256, 0, stream>>>(q_f, q_bf, H_Q,  (long)M_ROWS * H_Q  * 64);
  rope_cvt<<<4096,  256, 0, stream>>>(k_f, k_bf, H_KV, (long)M_ROWS * H_KV * 64);
  v_cvt_transpose<<<8192, 256, 0, stream>>>(v_f, vt_bf);
  attn_kernel<<<dim3(8, H_Q, B_SZ), 256, 0, stream>>>(q_bf, k_bf, vt_bf, y_f);
  cvt_bf16<<<32768, 256, 0, stream>>>(y_f, y_bf, (long)M_ROWS * DIMC);
  (void)hipMemcpyAsync(out, x, (size_t)M_ROWS * DIMC * sizeof(float),
                       hipMemcpyDeviceToDevice, stream);
  gemm_bf16<1><<<dim3(64, 16), 256, 0, stream>>>(y_bf, wo_t, out, M_ROWS, 2048, 2048);

  // FFN half (residual accumulated straight into d_out)
  rmsnorm_bf16<<<M_ROWS, 256, 0, stream>>>(out, fnw, a_bf);
  gemm_bf16<0><<<dim3(64, 64), 256, 0, stream>>>(a_bf, w1_t, h1, M_ROWS, HID, 2048);
  gemm_bf16<0><<<dim3(64, 64), 256, 0, stream>>>(a_bf, w2_t, h2, M_ROWS, HID, 2048);
  silu_mul_bf16<<<131072, 256, 0, stream>>>(h1, h2, hh, (long)M_ROWS * HID);
  gemm_bf16<1><<<dim3(64, 16), 256, 0, stream>>>(hh, w3_t, out, M_ROWS, 2048, HID);
}
